// MambaBaseline_29222957482427
// MI455X (gfx1250) — matrix operand order repaired and checked by reading, not compiled, not run on hardware
//
#include <hip/hip_runtime.h>

// ---------------------------------------------------------------------------
// Types for CDNA5 WMMA (wave32): bf16 in, fp32 accumulate
// ---------------------------------------------------------------------------
typedef __bf16 bf16;
typedef __attribute__((ext_vector_type(16))) __bf16 v16bf;
typedef __attribute__((ext_vector_type(8)))  __bf16 v8bf;
typedef __attribute__((ext_vector_type(8)))  float  v8f;

#define D_MODEL 768
#define D_INNER 1536
#define D_STATE 16
#define D_CONV  4
#define DT_RANK 48
#define XDBL_W  (DT_RANK + 2 * D_STATE)   // 80
#define BATCH   2
#define SEQ     2048
#define NTOK    (BATCH * SEQ)             // 4096
#define VOCAB   32000
#define NLAYERS 2
#define NCHUNK  16
#define CLEN    (SEQ / NCHUNK)            // 128

// ---------------------------------------------------------------------------
// f32 -> bf16 conversion (weights / activations)
// ---------------------------------------------------------------------------
__global__ __launch_bounds__(256) void cvt_f32_bf16(const float* __restrict__ in,
                                                    bf16* __restrict__ out, int n) {
    int i = blockIdx.x * 256 + threadIdx.x;
    if (i < n) out[i] = (bf16)in[i];
}

// W_dt [D_INNER][DT_RANK] -> [DT_RANK][D_INNER]  (coalesced dt_proj reads)
__global__ __launch_bounds__(256) void transpose_wdt(const float* __restrict__ W,
                                                     float* __restrict__ WT) {
    int i = blockIdx.x * 256 + threadIdx.x;
    if (i < D_INNER * DT_RANK) {
        int d = i / DT_RANK, r = i - d * DT_RANK;
        WT[r * D_INNER + d] = W[i];
    }
}

// ---------------------------------------------------------------------------
// Embedding gather: h[t,:] = emb[x[t],:]   (writes f32 + bf16 copies)
// ---------------------------------------------------------------------------
__global__ __launch_bounds__(256) void embed_kernel(const int* __restrict__ x,
                                                    const float* __restrict__ emb,
                                                    float* __restrict__ h,
                                                    bf16* __restrict__ hbf) {
    int t = blockIdx.x;
    int tok = x[t];
    const float* e = emb + (size_t)tok * D_MODEL;
    size_t base = (size_t)t * D_MODEL;
#pragma unroll
    for (int j = 0; j < 3; ++j) {
        int i = threadIdx.x + j * 256;
        float v = e[i];
        h[base + i] = v;
        hbf[base + i] = (bf16)v;
    }
}

// ---------------------------------------------------------------------------
// Generic bf16 WMMA GEMM:  C[M,N] = A[M,K] * W[N,K]^T  (+bias +residual)
// 256 threads = 8 waves; block tile 128x128; wave = 32 rows x 64 cols =
// 2 A-frags x 4 B-frags -> 8 independent v_wmma accumulators per K-chunk.
// Out-of-range columns are clamped to a valid weight row (results discarded
// at store time) so EXEC stays all-1s and the K loop is branch-free.
// ---------------------------------------------------------------------------
__global__ __launch_bounds__(256)
void gemm_bf16_wmma(const bf16* __restrict__ A, const bf16* __restrict__ W,
                    const float* __restrict__ bias, const float* __restrict__ residual,
                    float* __restrict__ C, bf16* __restrict__ Cbf,
                    int M, int N, int K, int nontemporal) {
    const int wave = threadIdx.x >> 5;
    const int lane = threadIdx.x & 31;
    const int half = lane >> 4;            // 0 or 1
    const int l16  = lane & 15;

    const int m0 = blockIdx.y * 128 + (wave >> 1) * 32;   // 2 A-frags: m0, m0+16
    const int n0 = blockIdx.x * 128 + (wave & 1) * 64;    // 4 B-tiles of 16 cols

    const int aoff = half << 4;            // A k-offset: 0 or 16
    const int boff = half << 4;            // B k-offset: 0 or 16

    const bf16* Ap0 = A + (size_t)(m0 + l16) * K + aoff;
    const bf16* Ap1 = Ap0 + (size_t)16 * K;

    const bf16* Wp[4];
#pragma unroll
    for (int j = 0; j < 4; ++j) {
        int col = n0 + j * 16 + l16;
        int ccol = (col < N) ? col : (N - 1);   // clamp: keep loads in-bounds
        Wp[j] = W + (size_t)ccol * K + boff;
    }

    v8f acc[2][4];
#pragma unroll
    for (int i = 0; i < 2; ++i)
#pragma unroll
        for (int j = 0; j < 4; ++j) acc[i][j] = (v8f){};

    for (int kk = 0; kk < K; kk += 32) {
        v8bf a0lo = *(const v8bf*)(Ap0 + kk);
        v8bf a0hi = *(const v8bf*)(Ap0 + kk + 8);
        v8bf a1lo = *(const v8bf*)(Ap1 + kk);
        v8bf a1hi = *(const v8bf*)(Ap1 + kk + 8);
        v16bf a0 = __builtin_shufflevector(a0lo, a0hi,
                     0,1,2,3,4,5,6,7,8,9,10,11,12,13,14,15);
        v16bf a1 = __builtin_shufflevector(a1lo, a1hi,
                     0,1,2,3,4,5,6,7,8,9,10,11,12,13,14,15);
        v16bf b[4];
#pragma unroll
        for (int j = 0; j < 4; ++j) b[j] = *(const v16bf*)(Wp[j] + kk);
#pragma unroll
        for (int j = 0; j < 4; ++j)
            acc[0][j] = __builtin_amdgcn_wmma_f32_16x16x32_bf16(
                false, a0, false, b[j], (short)0, acc[0][j], false, false);
#pragma unroll
        for (int j = 0; j < 4; ++j)
            acc[1][j] = __builtin_amdgcn_wmma_f32_16x16x32_bf16(
                false, a1, false, b[j], (short)0, acc[1][j], false, false);
    }

    // ---- store (N is always a multiple of 16 -> wave-uniform guards) ----
#pragma unroll
    for (int j = 0; j < 4; ++j) {
        int col = n0 + j * 16 + l16;
        if (col >= N) continue;
        float bv = bias ? bias[col] : 0.0f;
#pragma unroll
        for (int i = 0; i < 2; ++i) {
            int mb = m0 + i * 16 + (half << 3);
#pragma unroll
            for (int r = 0; r < 8; ++r) {
                size_t idx = (size_t)(mb + r) * N + col;
                float v = acc[i][j][r] + bv;
                if (residual) v += residual[idx];
                if (nontemporal) __builtin_nontemporal_store(v, &C[idx]);
                else             C[idx] = v;
                if (Cbf) Cbf[idx] = (bf16)v;
            }
        }
    }
}

// ---------------------------------------------------------------------------
// Causal depthwise conv (width 4) + SiLU.  Reads u_pre = xz[:, :D_INNER].
// ---------------------------------------------------------------------------
__global__ __launch_bounds__(256)
void conv_silu_kernel(const float* __restrict__ xz, const float* __restrict__ cw,
                      const float* __restrict__ cb, float* __restrict__ u,
                      bf16* __restrict__ ubf) {
    int idx = blockIdx.x * 256 + threadIdx.x;      // [0, NTOK*D_INNER)
    int t = idx / D_INNER;
    int c = idx - t * D_INNER;
    int b = t >> 11, l = t & (SEQ - 1);
    float acc = cb[c];
#pragma unroll
    for (int j = 0; j < D_CONV; ++j) {
        int ls = l - (D_CONV - 1) + j;
        if (ls >= 0)
            acc += xz[(size_t)(b * SEQ + ls) * (2 * D_INNER) + c] * cw[c * D_CONV + j];
    }
    float s = acc / (1.0f + expf(-acc));           // SiLU
    u[idx] = s;
    ubf[idx] = (bf16)s;
}

// ---------------------------------------------------------------------------
// dt = softplus(dtl @ W_dt^T + b_dt)   (K=48; WdtT is [r][d] -> coalesced)
// ---------------------------------------------------------------------------
__global__ __launch_bounds__(256)
void dt_proj_kernel(const float* __restrict__ xdbl, const float* __restrict__ WdtT,
                    const float* __restrict__ bdt, float* __restrict__ dtb) {
    int idx = blockIdx.x * 256 + threadIdx.x;      // [0, NTOK*D_INNER)
    int t = idx / D_INNER;
    int d = idx - t * D_INNER;
    const float* xr = xdbl + (size_t)t * XDBL_W;
    float acc = bdt[d];
#pragma unroll 8
    for (int r = 0; r < DT_RANK; ++r) acc += xr[r] * WdtT[r * D_INNER + d];
    dtb[idx] = (acc > 20.0f) ? acc : log1pf(expf(acc));
}

// ---------------------------------------------------------------------------
// Chunked selective scan.  h_t = a_t*h_{t-1} + b_t is a linear recurrence:
//   phase 1: per (b,d,s,chunk) compute transfer P=prod(a), Q=state-from-zero
//   phase 2: per (b,d,s,chunk) combine preceding (P,Q) -> true initial state,
//            re-scan chunk, butterfly-reduce 16 states into y (shfl_xor),
//            fuse D_skip and SiLU(z) gate.
// Serial depth drops 2048 -> ~CLEN+NCHUNK, parallelism rises 16x to 786K lanes.
// gid decode keeps the 16 s-lanes of one (pair,chunk) contiguous in a wave
// half so the xor masks 1/2/4/8 stay inside the group.
// ---------------------------------------------------------------------------
__global__ __launch_bounds__(256)
void scan_phase1(const float* __restrict__ dt, const float* __restrict__ u,
                 const float* __restrict__ xdbl, const float* __restrict__ A_log,
                 float* __restrict__ Pc, float* __restrict__ Qc) {
    int gid = blockIdx.x * 256 + threadIdx.x;   // [0, BATCH*D_INNER*16*NCHUNK)
    int s = gid & (D_STATE - 1);
    int rest = gid >> 4;
    int c = rest & (NCHUNK - 1);
    int pair = rest >> 4;                       // (b, d)
    int d = pair % D_INNER;
    int b = pair / D_INNER;

    const float A = -expf(A_log[d * D_STATE + s]);
    float P = 1.0f, hs = 0.0f;
    int t0 = b * SEQ + c * CLEN;
    for (int i = 0; i < CLEN; ++i) {
        int t = t0 + i;
        float dtv = dt[(size_t)t * D_INNER + d];
        float a = expf(dtv * A);
        float bb = (dtv * u[(size_t)t * D_INNER + d]) * xdbl[(size_t)t * XDBL_W + DT_RANK + s];
        hs = a * hs + bb;
        P *= a;
    }
    size_t o = ((size_t)pair * D_STATE + s) * NCHUNK + c;
    Pc[o] = P;
    Qc[o] = hs;
}

__global__ __launch_bounds__(256)
void scan_phase2(const float* __restrict__ dt, const float* __restrict__ u,
                 const float* __restrict__ xdbl, const float* __restrict__ xz,
                 const float* __restrict__ A_log, const float* __restrict__ D_skip,
                 const float* __restrict__ Pc, const float* __restrict__ Qc,
                 bf16* __restrict__ ybf) {
    int gid = blockIdx.x * 256 + threadIdx.x;   // [0, BATCH*D_INNER*16*NCHUNK)
    int s = gid & (D_STATE - 1);
    int rest = gid >> 4;
    int c = rest & (NCHUNK - 1);
    int pair = rest >> 4;
    int d = pair % D_INNER;
    int b = pair / D_INNER;

    // combine preceding chunk transfers -> initial state for this chunk
    size_t base = ((size_t)pair * D_STATE + s) * NCHUNK;
    float hs = 0.0f;
    for (int j = 0; j < c; ++j) hs = Pc[base + j] * hs + Qc[base + j];

    const float A   = -expf(A_log[d * D_STATE + s]);
    const float Dsk = D_skip[d];
    int t0 = b * SEQ + c * CLEN;
    for (int i = 0; i < CLEN; ++i) {
        int t = t0 + i;
        float dtv = dt[(size_t)t * D_INNER + d];
        float uv  = u[(size_t)t * D_INNER + d];
        float Bv  = xdbl[(size_t)t * XDBL_W + DT_RANK + s];
        float Cv  = xdbl[(size_t)t * XDBL_W + DT_RANK + D_STATE + s];
        hs = expf(dtv * A) * hs + (dtv * uv) * Bv;
        float p = hs * Cv;
        p += __shfl_xor(p, 1, 32);
        p += __shfl_xor(p, 2, 32);
        p += __shfl_xor(p, 4, 32);
        p += __shfl_xor(p, 8, 32);
        if (s == 0) {
            float zv = xz[(size_t)t * (2 * D_INNER) + D_INNER + d];
            float y = (p + uv * Dsk) * (zv / (1.0f + expf(-zv)));
            ybf[(size_t)t * D_INNER + d] = (bf16)y;
        }
    }
}

// ---------------------------------------------------------------------------
// LayerNorm over D_MODEL=768 (block per token, LDS tree reduction) -> bf16
// ---------------------------------------------------------------------------
__global__ __launch_bounds__(256)
void layernorm_kernel(const float* __restrict__ h, const float* __restrict__ g,
                      const float* __restrict__ bb, bf16* __restrict__ out) {
    __shared__ float ssum[256];
    __shared__ float ssq[256];
    int t = blockIdx.x;
    const float* hp = h + (size_t)t * D_MODEL;
    float x0 = hp[threadIdx.x];
    float x1 = hp[threadIdx.x + 256];
    float x2 = hp[threadIdx.x + 512];
    ssum[threadIdx.x] = x0 + x1 + x2;
    ssq[threadIdx.x]  = x0 * x0 + x1 * x1 + x2 * x2;
    __syncthreads();
    for (int o = 128; o > 0; o >>= 1) {
        if (threadIdx.x < o) {
            ssum[threadIdx.x] += ssum[threadIdx.x + o];
            ssq[threadIdx.x]  += ssq[threadIdx.x + o];
        }
        __syncthreads();
    }
    float mu  = ssum[0] * (1.0f / D_MODEL);
    float var = ssq[0] * (1.0f / D_MODEL) - mu * mu;
    float inv = rsqrtf(var + 1e-5f);
    bf16* op = out + (size_t)t * D_MODEL;
    op[threadIdx.x]       = (bf16)((x0 - mu) * inv * g[threadIdx.x]       + bb[threadIdx.x]);
    op[threadIdx.x + 256] = (bf16)((x1 - mu) * inv * g[threadIdx.x + 256] + bb[threadIdx.x + 256]);
    op[threadIdx.x + 512] = (bf16)((x2 - mu) * inv * g[threadIdx.x + 512] + bb[threadIdx.x + 512]);
}

// ---------------------------------------------------------------------------
// Host-side orchestration
// ---------------------------------------------------------------------------
static inline size_t ws_alloc(size_t* off, size_t bytes) {
    size_t p = *off;
    *off = (p + bytes + 255) & ~(size_t)255;
    return p;
}

extern "C" void kernel_launch(void* const* d_in, const int* in_sizes, int n_in,
                              void* d_out, int out_size, void* d_ws, size_t ws_size,
                              hipStream_t stream) {
    (void)in_sizes; (void)n_in; (void)out_size; (void)ws_size;

    const int*   x      = (const int*)  d_in[0];
    const float* emb    = (const float*)d_in[1];
    const float* W_in   = (const float*)d_in[2];
    const float* conv_w = (const float*)d_in[3];
    const float* conv_b = (const float*)d_in[4];
    const float* W_x    = (const float*)d_in[5];
    const float* W_dt   = (const float*)d_in[6];
    const float* b_dt   = (const float*)d_in[7];
    const float* A_log  = (const float*)d_in[8];
    const float* D_skip = (const float*)d_in[9];
    const float* W_out  = (const float*)d_in[10];
    const float* ln_g   = (const float*)d_in[11];
    const float* ln_b   = (const float*)d_in[12];
    const float* head_w = (const float*)d_in[13];
    const float* head_b = (const float*)d_in[14];
    float* out = (float*)d_out;

    char* ws = (char*)d_ws;
    size_t off = 0;
    float* h      = (float*)(ws + ws_alloc(&off, (size_t)NTOK * D_MODEL * 4));
    bf16*  hbf    = (bf16*) (ws + ws_alloc(&off, (size_t)NTOK * D_MODEL * 2));
    float* xz     = (float*)(ws + ws_alloc(&off, (size_t)NTOK * 2 * D_INNER * 4));
    float* u      = (float*)(ws + ws_alloc(&off, (size_t)NTOK * D_INNER * 4));
    bf16*  ubf    = (bf16*) (ws + ws_alloc(&off, (size_t)NTOK * D_INNER * 2));
    float* xdbl   = (float*)(ws + ws_alloc(&off, (size_t)NTOK * XDBL_W * 4));
    float* dtb    = (float*)(ws + ws_alloc(&off, (size_t)NTOK * D_INNER * 4));
    bf16*  ybf    = (bf16*) (ws + ws_alloc(&off, (size_t)NTOK * D_INNER * 2));
    bf16*  winbf  = (bf16*) (ws + ws_alloc(&off, (size_t)2 * D_INNER * D_MODEL * 2));
    bf16*  wxbf   = (bf16*) (ws + ws_alloc(&off, (size_t)XDBL_W * D_INNER * 2));
    bf16*  woutbf = (bf16*) (ws + ws_alloc(&off, (size_t)D_MODEL * D_INNER * 2));
    float* wdtT   = (float*)(ws + ws_alloc(&off, (size_t)DT_RANK * D_INNER * 4));
    float* Pc     = (float*)(ws + ws_alloc(&off, (size_t)BATCH * D_INNER * D_STATE * NCHUNK * 4));
    float* Qc     = (float*)(ws + ws_alloc(&off, (size_t)BATCH * D_INNER * D_STATE * NCHUNK * 4));
    bf16*  hnbf   = (bf16*) (ws + ws_alloc(&off, (size_t)NTOK * D_MODEL * 2));
    bf16*  headbf = (bf16*) (ws + ws_alloc(&off, (size_t)VOCAB * D_MODEL * 2));

    const int EW = NTOK * D_INNER;                  // elementwise work size
    const int SCAN_T = BATCH * D_INNER * D_STATE * NCHUNK;  // 786432
    const dim3 blk(256);

    // Embedding
    embed_kernel<<<NTOK, blk, 0, stream>>>(x, emb, h, hbf);

    for (int l = 0; l < NLAYERS; ++l) {
        const float* Wl_in  = W_in   + (size_t)l * 2 * D_INNER * D_MODEL;
        const float* cwl    = conv_w + (size_t)l * D_INNER * D_CONV;
        const float* cbl    = conv_b + (size_t)l * D_INNER;
        const float* Wl_x   = W_x    + (size_t)l * XDBL_W * D_INNER;
        const float* Wl_dt  = W_dt   + (size_t)l * D_INNER * DT_RANK;
        const float* bl_dt  = b_dt   + (size_t)l * D_INNER;
        const float* Al_log = A_log  + (size_t)l * D_INNER * D_STATE;
        const float* Dl     = D_skip + (size_t)l * D_INNER;
        const float* Wl_out = W_out  + (size_t)l * D_MODEL * D_INNER;

        // Convert layer weights to bf16 (+ transpose W_dt)
        {
            int n = 2 * D_INNER * D_MODEL;
            cvt_f32_bf16<<<(n + 255) / 256, blk, 0, stream>>>(Wl_in, winbf, n);
            n = XDBL_W * D_INNER;
            cvt_f32_bf16<<<(n + 255) / 256, blk, 0, stream>>>(Wl_x, wxbf, n);
            n = D_MODEL * D_INNER;
            cvt_f32_bf16<<<(n + 255) / 256, blk, 0, stream>>>(Wl_out, woutbf, n);
            n = D_INNER * DT_RANK;
            transpose_wdt<<<(n + 255) / 256, blk, 0, stream>>>(Wl_dt, wdtT);
        }

        // in_proj: xz[NTOK, 3072] = h @ W_in^T
        {
            dim3 g((2 * D_INNER + 127) / 128, NTOK / 128);
            gemm_bf16_wmma<<<g, blk, 0, stream>>>(hbf, winbf, nullptr, nullptr,
                                                  xz, nullptr, NTOK, 2 * D_INNER, D_MODEL, 0);
        }

        // depthwise causal conv + SiLU -> u, ubf
        conv_silu_kernel<<<EW / 256, blk, 0, stream>>>(xz, cwl, cbl, u, ubf);

        // x_proj: xdbl[NTOK, 80] = u @ W_x^T
        {
            dim3 g((XDBL_W + 127) / 128, NTOK / 128);
            gemm_bf16_wmma<<<g, blk, 0, stream>>>(ubf, wxbf, nullptr, nullptr,
                                                  xdbl, nullptr, NTOK, XDBL_W, D_INNER, 0);
        }

        // dt_proj + softplus
        dt_proj_kernel<<<EW / 256, blk, 0, stream>>>(xdbl, wdtT, bl_dt, dtb);

        // chunk-parallel selective scan (fused D_skip + SiLU(z) gate) -> ybf
        scan_phase1<<<SCAN_T / 256, blk, 0, stream>>>(dtb, u, xdbl, Al_log, Pc, Qc);
        scan_phase2<<<SCAN_T / 256, blk, 0, stream>>>(dtb, u, xdbl, xz, Al_log, Dl,
                                                      Pc, Qc, ybf);

        // out_proj + residual: h = h + y @ W_out^T  (also refresh hbf)
        {
            dim3 g((D_MODEL + 127) / 128, NTOK / 128);
            gemm_bf16_wmma<<<g, blk, 0, stream>>>(ybf, woutbf, nullptr, h,
                                                  h, hbf, NTOK, D_MODEL, D_INNER, 0);
        }
    }

    // Final LayerNorm -> hnbf
    layernorm_kernel<<<NTOK, blk, 0, stream>>>(h, ln_g, ln_b, hnbf);

    // LM head: out[NTOK, VOCAB] = hn @ head_w^T + head_b  (nontemporal stores)
    {
        int n = VOCAB * D_MODEL;
        cvt_f32_bf16<<<(n + 255) / 256, blk, 0, stream>>>(head_w, headbf, n);
        dim3 g(VOCAB / 128, NTOK / 128);
        gemm_bf16_wmma<<<g, blk, 0, stream>>>(hnbf, headbf, head_b, nullptr,
                                              out, nullptr, NTOK, VOCAB, D_MODEL, 1);
    }
}
